// SplatGaussian2D_73031623901490
// MI455X (gfx1250) — matrix operand order repaired and checked by reading, not compiled, not run on hardware
//
#include <hip/hip_runtime.h>

// ---------------------------------------------------------------------------
// SplatGaussian2D for MI455X (gfx1250, wave32, WMMA)
//
// out[b,c] = sum_n exp(-|diag(S_n) R(a_n) (p_b - mu_n)|^2) * sigmoid(rgb)[n,c]
//
// Strategy:
//   prep kernel (N=2048): fold tanh/sincos/sigmoid into per-gaussian
//     {mx,my,a00,a01,a10,a11} with the 2x2 matrix pre-scaled by sqrt(log2 e)
//     so the weight is exp2(-(dx^2+dy^2)) -> one v_exp_f32 with neg modifier.
//     Also build rgbT[16][N] f16 (rows 3..15 zero) = A-matrix source.
//   main kernel: wave handles 16 pixels x all 2048 gaussians.
//     Per 32-gaussian chunk: lane computes 16 f16 weights (B fragment,
//     32x16, lane half h covers K = h*16+e), loads the rgb^T A fragment
//     (16x32), and accumulates with v_wmma_f32_16x16x32_f16 into f32 C.
//     D rows 0..2 are the RGB channels: lane l<16 stores pixel l directly.
// ---------------------------------------------------------------------------

typedef __attribute__((ext_vector_type(16))) _Float16 v16h;
typedef __attribute__((ext_vector_type(8)))  _Float16 v8h;
typedef __attribute__((ext_vector_type(8)))  float    v8f;

#define NG   2048          // gaussians (compile-time; matches reference)
#define WPB  8             // waves per block
#define TPB  (WPB * 32)    // 256 threads
#define PIX_PER_BLOCK (WPB * 16)   // 128 pixels per block

__global__ __launch_bounds__(TPB) void splat_prep(
    const float* __restrict__ rgb,   const float* __restrict__ mu,
    const float* __restrict__ scale, const float* __restrict__ angle,
    float4* __restrict__ geo, _Float16* __restrict__ rgbT, int n)
{
    int g = blockIdx.x * blockDim.x + threadIdx.x;
    if (g >= n) return;

    const float L2E = 1.44269504089f;       // log2(e)
    const float KS  = 1.20112240872f;       // sqrt(log2 e), folded into matrix
    const float SMIN = 1.0f / 30.0f, SMAX = 1.0f / 0.75f;

    // mu_t * 0.5*wh = tanh(mu) * 1.05 * 256   (W=H=512)
    float mx = tanhf(mu[2*g+0]) * (1.05f * 256.0f);
    float my = tanhf(mu[2*g+1]) * (1.05f * 256.0f);

    float al = tanhf(angle[g]) * 3.1416f;
    float c = __cosf(al), s = __sinf(al);

    float s0 = 1.0f / (1.0f + __builtin_amdgcn_exp2f(-scale[2*g+0] * L2E));
    float s1 = 1.0f / (1.0f + __builtin_amdgcn_exp2f(-scale[2*g+1] * L2E));
    float S0 = s0 * (SMAX - SMIN) + SMIN;
    float S1 = s1 * (SMAX - SMIN) + SMIN;

    // d1x = a00*vx - a01*vy ; d1y = a10*vx + a11*vy  (pre-scaled by KS)
    geo[2*g+0] = make_float4(mx, my, KS * S0 * c, KS * S0 * s);
    geo[2*g+1] = make_float4(KS * S1 * s, KS * S1 * c, 0.0f, 0.0f);

    // rgb^T in f16, padded to 16 channel rows (rows 3..15 zero) so the
    // WMMA A-matrix needs no masking.
#pragma unroll
    for (int ch = 0; ch < 16; ++ch) {
        float v = 0.0f;
        if (ch < 3)
            v = 1.0f / (1.0f + __builtin_amdgcn_exp2f(-rgb[3*g+ch] * L2E));
        rgbT[ch * n + g] = (_Float16)v;
    }
}

__global__ __launch_bounds__(TPB) void splat_main(
    const float*  __restrict__ x,      // [B,2] pixel coords
    const float4* __restrict__ geo,    // [NG][2] float4
    const v8h*    __restrict__ rgbT,   // 16 rows x NG/8 v8h (row-major)
    float*        __restrict__ out)    // [B,3]
{
    __shared__ float4 sgeo[NG * 2];    // 64 KB staged gaussian params
    for (int i = threadIdx.x; i < NG * 2; i += TPB) sgeo[i] = geo[i];
    __syncthreads();

    const int lane = threadIdx.x & 31;
    const int wave = threadIdx.x >> 5;
    const int col  = lane & 15;        // pixel column (B/N dim) == A row (channel)
    const int h    = lane >> 4;        // lane half

    const int pix = blockIdx.x * PIX_PER_BLOCK + wave * 16 + col;
    // (xn - mu_t)*0.5*wh  ==  (x - 256) - tanh(mu)*1.05*256
    const float px = x[2*pix+0] - 256.0f;
    const float py = x[2*pix+1] - 256.0f;

    v8f acc = {};                      // 16x16 f32 C/D accumulator
    const int arow = col * (NG / 8);   // rgbT row base, in v8h units

    for (int ck = 0; ck < NG / 32; ++ck) {
        // -------- A fragment: rgb^T 16x32 f16 --------
        // A k order used here (matches B): lane half h holds K = h*16 + e
        union { v16h v; v8h p[2]; } a;
        a.p[0] = rgbT[arow + ck * 4 + 2 * h];        // K = h*16 .. h*16+7
        a.p[1] = rgbT[arow + ck * 4 + 2 * h + 1];    // K = h*16+8 .. h*16+15

        // -------- B fragment: weight tile 32x16 f16 --------
        // B k order used here: lane half h holds K = h*16 + e for column col.
        v16h b;
        const int g0 = ck * 32 + h * 16;
#pragma unroll
        for (int e = 0; e < 16; ++e) {
            float4 q0 = sgeo[(g0 + e) * 2 + 0];  // mx,my,a00,a01
            float4 q1 = sgeo[(g0 + e) * 2 + 1];  // a10,a11,-,-
            float vx = px - q0.x;
            float vy = py - q0.y;
            float dx = q0.z * vx - q0.w * vy;
            float dy = q1.x * vx + q1.y * vy;
            float r2 = dx * dx + dy * dy;        // already scaled by log2(e)
            b[e] = (_Float16)__builtin_amdgcn_exp2f(-r2);
        }

        // D(16x16,f32) += A(16x32,f16) x B(32x16,f16)
        acc = __builtin_amdgcn_wmma_f32_16x16x32_f16(
            /*neg_a=*/false, a.v, /*neg_b=*/false, b,
            /*c_mod=*/(short)0, acc, /*reuse_a=*/false, /*reuse_b=*/false);
    }

    // D layout: VGPR j, lanes 0..15 -> M=j (channel), N=lane (pixel).
    if (lane < 16) {
        const int p = blockIdx.x * PIX_PER_BLOCK + wave * 16 + lane;
        out[3*p + 0] = acc[0];
        out[3*p + 1] = acc[1];
        out[3*p + 2] = acc[2];
    }
}

extern "C" void kernel_launch(void* const* d_in, const int* in_sizes, int n_in,
                              void* d_out, int out_size, void* d_ws, size_t ws_size,
                              hipStream_t stream)
{
    const float* x     = (const float*)d_in[0];   // [B,2]
    const float* rgb   = (const float*)d_in[1];   // [N,3]
    const float* mu    = (const float*)d_in[2];   // [N,2]
    const float* scale = (const float*)d_in[3];   // [N,2]
    const float* angle = (const float*)d_in[4];   // [N]

    const int N    = in_sizes[4];        // 2048
    const int Bpix = in_sizes[0] / 2;    // 65536

    // workspace: geo (N*32B = 64KB) | rgbT f16 padded 16 rows (64KB)
    float4*   geo  = (float4*)d_ws;
    _Float16* rgbT = (_Float16*)((char*)d_ws + (size_t)N * 2 * sizeof(float4));

    splat_prep<<<(N + TPB - 1) / TPB, TPB, 0, stream>>>(rgb, mu, scale, angle,
                                                        geo, rgbT, N);
    splat_main<<<Bpix / PIX_PER_BLOCK, TPB, 0, stream>>>(x, geo,
                                                         (const v8h*)rgbT,
                                                         (float*)d_out);
}
